// NeuralODE_57999238365256
// MI455X (gfx1250) — matrix operand order repaired and checked by reading, not compiled, not run on hardware
//
#include <hip/hip_runtime.h>
#include <math.h>

// ---------------- CDNA5 / problem constants ----------------
typedef __attribute__((ext_vector_type(16))) __bf16 v16bf;
typedef __attribute__((ext_vector_type(8)))  float  v8f;

#define NWG   24          // persistent workgroups (each = 8 wave32)
#define TPB   256
#define GSZ   (NWG * TPB)
#define NWAVES (GSZ / 32)

#define DATA_DIM 768
#define HIDDEN   768
#define STATE    3072
#define NSTEPS   100

// workspace layout (bytes)
static constexpr size_t OFF_SYNC = 0;                      // 2 u32: count, generation
static constexpr size_t OFF_PART = 256;                    // per-WG error partials
static constexpr size_t OFF_Y    = 1024;
static constexpr size_t VEC_B    = STATE * 4;              // 12288
static constexpr size_t OFF_Y1   = OFF_Y  + VEC_B;
static constexpr size_t OFF_X    = OFF_Y1 + VEC_B;
static constexpr size_t OFF_K    = OFF_X  + VEC_B;         // 7 k-vectors
static constexpr size_t OFF_HA   = OFF_K + 7 * VEC_B;
static constexpr size_t OFF_HB   = OFF_HA + 4096;
// packed weights: per 16x32 tile -> 512 hi bf16 + 512 lo bf16 = 2048 B
static constexpr size_t OFF_WIN  = 262144;
static constexpr size_t WIN_B    = (size_t)(HIDDEN/16) * (STATE/32) * 2048;   // 9.4 MB
static constexpr size_t OFF_WH   = OFF_WIN + WIN_B;
static constexpr size_t WH_STRIDE= (size_t)(HIDDEN/16) * (HIDDEN/32) * 2048;  // 2.36 MB
static constexpr size_t OFF_WOUT = OFF_WH + 3 * WH_STRIDE;
static constexpr size_t WOUT_B   = (size_t)(STATE/16) * (HIDDEN/32) * 2048;   // 9.4 MB

// Tsit5 tableau
__constant__ float AC[5][5] = {
  {0.161f, 0.f, 0.f, 0.f, 0.f},
  {-0.008480655492356989f, 0.335480655492357f, 0.f, 0.f, 0.f},
  {2.8971530571054935f, -6.359448489975075f, 4.3622954328695815f, 0.f, 0.f},
  {5.325864828439257f, -11.748883564062828f, 7.4955393428898365f, -0.09249506636175525f, 0.f},
  {5.86145544294642f, -12.92096931784711f, 8.159367898576159f, -0.071584973281401f, -0.028269050394068383f}};
__constant__ float BC[6] = {0.09646076681806523f, 0.01f, 0.4798896504144996f,
                            1.379008574103742f, -3.290069515436081f, 2.324710524099774f};
__constant__ float EC[7] = {-0.001780011052226f, -0.000816434459657f, 0.007880878010262f,
                            -0.144711007173263f, 0.582357165452555f, -0.458082105929187f,
                            0.015151515151515152f};

// ---------------- prep kernels ----------------
// Pack W (row-major N x K, f32) into WMMA B-operand tiles, split hi/lo bf16.
// B layout used here (contiguous k, not AMD's table order): lane l holds column N=l&15; elems e=0..15 map to
// K = e + 16*(l>>4). Tile block: [0..511]=hi, [512..1023]=lo, addressed [lane*16+e].
__global__ void pack_kernel(const float* __restrict__ W, int N, int K,
                            char* __restrict__ ws, size_t dstOff) {
  __bf16* dst = (__bf16*)(ws + dstOff);
  int total = N * K;
  for (int idx = blockIdx.x * blockDim.x + threadIdx.x; idx < total;
       idx += gridDim.x * blockDim.x) {
    int n = idx / K, k = idx - n * K;
    float w = W[idx];
    __bf16 hi = (__bf16)w;
    __bf16 lo = (__bf16)(w - (float)hi);
    int nt = n >> 4, nn = n & 15, kt = k >> 5, kk = k & 31;
    int lane = nn + (kk & 16);
    int e = kk & 15;
    __bf16* tb = dst + ((size_t)nt * (K >> 5) + kt) * 1024;
    tb[lane * 16 + e]       = hi;
    tb[512 + lane * 16 + e] = lo;
  }
}

__global__ void init_sync(unsigned* s) {
  if (threadIdx.x == 0 && blockIdx.x == 0) { s[0] = 0u; s[1] = 0u; }
}

// ---------------- device helpers ----------------
__device__ __forceinline__ void gbar(unsigned* sync) {
  __threadfence();
  __syncthreads();
  if (threadIdx.x == 0) {
    volatile unsigned* vgen = sync + 1;
    unsigned g = *vgen;
    unsigned arrived = atomicAdd(sync, 1u) + 1u;
    if (arrived == (unsigned)NWG) {
      atomicExch(sync, 0u);
      __threadfence();
      atomicAdd(sync + 1, 1u);
    } else {
      while (*vgen == g) { __builtin_amdgcn_s_sleep(1); }
    }
  }
  __syncthreads();
  __threadfence();
}

__device__ __forceinline__ float block_sum(float v, float* red) {
  int tid = threadIdx.x;
  red[tid] = v; __syncthreads();
  for (int s = TPB / 2; s > 0; s >>= 1) {
    if (tid < s) red[tid] += red[tid + s];
    __syncthreads();
  }
  float r = red[0]; __syncthreads();
  return r;
}

// out[n] = act( sum_k W[n,k] x[k] + b[n] ), W pre-packed hi/lo bf16 tiles.
// Split product: W*x ~= Whi*xhi + Wlo*xhi + Whi*xlo (drop lo*lo).
__device__ __forceinline__ void matvec_wmma(const unsigned short* __restrict__ pwRaw,
                                            const float* __restrict__ bias,
                                            const float* __restrict__ x,
                                            float* __restrict__ outv,
                                            int NT, int KT, bool act,
                                            int wave, int lane) {
  const __bf16* pw = (const __bf16*)pwRaw;
  const v8f z = {0.f, 0.f, 0.f, 0.f, 0.f, 0.f, 0.f, 0.f};
  for (int nt = wave; nt < NT; nt += NWAVES) {
    v8f a0 = z, a1 = z, a2 = z;
    const __bf16* tbase = pw + (size_t)nt * KT * 1024;
    for (int kt = 0; kt < KT; ++kt) {
      const __bf16* tb = tbase + (size_t)kt * 1024;
      v16bf Bhi = *(const v16bf*)(tb + lane * 16);
      v16bf Blo = *(const v16bf*)(tb + 512 + lane * 16);
      // A row-broadcast: every lane supplies the same activation chunk; the
      // chosen (contiguous) K index for lane l, elem e is e+16*(l>>4) (matches packed B) --
      // matched here so row M=0 (the only row we read) sees x[kt*32 ..].
      const float* xs = x + kt * 32 + ((lane & 16) ? 16 : 0);
      v16bf Ahi, Alo;
#pragma unroll
      for (int j = 0; j < 8; ++j) {
        float v0 = xs[j], v1 = xs[8 + j];
        __bf16 h0 = (__bf16)v0, h1 = (__bf16)v1;
        Ahi[j]     = h0;  Alo[j]     = (__bf16)(v0 - (float)h0);
        Ahi[8 + j] = h1;  Alo[8 + j] = (__bf16)(v1 - (float)h1);
      }
      a0 = __builtin_amdgcn_wmma_f32_16x16x32_bf16(false, Ahi, false, Bhi, (short)0, a0, false, false);
      a1 = __builtin_amdgcn_wmma_f32_16x16x32_bf16(false, Alo, false, Bhi, (short)0, a1, false, false);
      a2 = __builtin_amdgcn_wmma_f32_16x16x32_bf16(false, Ahi, false, Blo, (short)0, a2, false, false);
    }
    if (lane < 16) {  // D row 0 lives in C/D VGPR0, lanes 0..15 (N = lane)
      int n = nt * 16 + lane;
      float v = a0[0] + a1[0] + a2[0] + bias[n];
      if (act) v = fmaxf(v, 0.f) + log1pf(expf(-fabsf(v)));  // softplus
      outv[n] = v;
    }
  }
}

struct Ctx {
  const unsigned short *pWin, *pWh0, *pWh1, *pWh2, *pWout;
  const float *b_in, *b_h0, *b_h1, *b_h2, *b_out;
  float *hA, *hB;
  unsigned* sync;
  int wave, lane;
};

__device__ __forceinline__ void evalf(const Ctx& c, const float* xin, float* kout) {
  matvec_wmma(c.pWin, c.b_in, xin,  c.hA,  HIDDEN/16, STATE/32,  true,  c.wave, c.lane); gbar(c.sync);
  matvec_wmma(c.pWh0, c.b_h0, c.hA, c.hB,  HIDDEN/16, HIDDEN/32, true,  c.wave, c.lane); gbar(c.sync);
  matvec_wmma(c.pWh1, c.b_h1, c.hB, c.hA,  HIDDEN/16, HIDDEN/32, true,  c.wave, c.lane); gbar(c.sync);
  matvec_wmma(c.pWh2, c.b_h2, c.hA, c.hB,  HIDDEN/16, HIDDEN/32, true,  c.wave, c.lane); gbar(c.sync);
  matvec_wmma(c.pWout, c.b_out, c.hB, kout, STATE/16, HIDDEN/32, false, c.wave, c.lane); gbar(c.sync);
}

// ---------------- main persistent kernel ----------------
__global__ void __launch_bounds__(TPB, 1)
ode_main(const float* __restrict__ y0, const float* __restrict__ b_in,
         const float* __restrict__ b_hid, const float* __restrict__ b_out,
         const float* __restrict__ eps, float* __restrict__ out,
         char* __restrict__ ws) {
  unsigned* sync = (unsigned*)(ws + OFF_SYNC);
  float* part = (float*)(ws + OFF_PART);
  float* yv = (float*)(ws + OFF_Y);
  float* y1 = (float*)(ws + OFF_Y1);
  float* xv = (float*)(ws + OFF_X);
  float* kv[7];
#pragma unroll
  for (int j = 0; j < 7; ++j) kv[j] = (float*)(ws + OFF_K + (size_t)j * VEC_B);

  Ctx c;
  c.pWin = (const unsigned short*)(ws + OFF_WIN);
  c.pWh0 = (const unsigned short*)(ws + OFF_WH);
  c.pWh1 = (const unsigned short*)(ws + OFF_WH + WH_STRIDE);
  c.pWh2 = (const unsigned short*)(ws + OFF_WH + 2 * WH_STRIDE);
  c.pWout = (const unsigned short*)(ws + OFF_WOUT);
  c.b_in = b_in; c.b_h0 = b_hid; c.b_h1 = b_hid + HIDDEN; c.b_h2 = b_hid + 2 * HIDDEN;
  c.b_out = b_out;
  c.hA = (float*)(ws + OFF_HA);
  c.hB = (float*)(ws + OFF_HB);
  c.sync = sync;

  int tid = threadIdx.x;
  int gtid = blockIdx.x * TPB + tid;
  c.lane = tid & 31;
  c.wave = gtid >> 5;
  __shared__ float red[TPB];

  // init state + output row 0
  for (int i = gtid; i < STATE; i += GSZ) yv[i] = y0[i];
  for (int i = gtid; i < DATA_DIM; i += GSZ)
    out[i] = y0[i] + eps[i] * y0[DATA_DIM + i];
  gbar(sync);

  float t = 0.f, dtv = 0.1f;   // DT0
  bool havek1 = false;         // Tsit5 is FSAL: k1 only changes on accept (-> k7)
  for (int step = 1; step < NSTEPS; ++step) {
    float t1 = (float)step * (1.0f / 99.0f);
    for (int att = 0; att < 6; ++att) {
      if (t >= t1 - 1e-10f) break;      // "done" attempts are exact no-ops
      float dte = fminf(dtv, t1 - t);

      if (!havek1) { evalf(c, yv, kv[0]); havek1 = true; }

      // stages 2..6
      for (int st = 0; st < 5; ++st) {
        for (int i = gtid; i < STATE; i += GSZ) {
          float s = 0.f;
          for (int j = 0; j <= st; ++j) s += AC[st][j] * kv[j][i];
          xv[i] = yv[i] + dte * s;
        }
        gbar(sync);
        evalf(c, xv, kv[st + 1]);
      }
      // y1 and k7
      for (int i = gtid; i < STATE; i += GSZ) {
        float s = 0.f;
        for (int j = 0; j < 6; ++j) s += BC[j] * kv[j][i];
        y1[i] = yv[i] + dte * s;
      }
      gbar(sync);
      evalf(c, y1, kv[6]);

      // error norm
      float acc = 0.f;
      for (int i = gtid; i < STATE; i += GSZ) {
        float e = 0.f;
        for (int j = 0; j < 7; ++j) e += EC[j] * kv[j][i];
        e *= dte;
        float sc = 1e-6f + 1e-3f * fmaxf(fabsf(yv[i]), fabsf(y1[i]));
        float r = e / sc;
        acc += r * r;
      }
      float bs = block_sum(acc, red);
      if (tid == 0) part[blockIdx.x] = bs;
      gbar(sync);
      float tot = 0.f;
      for (int w = 0; w < NWG; ++w) tot += part[w];
      float err = sqrtf(tot * (1.0f / (float)STATE));

      bool accept = (err <= 1.0f);
      float factor = fminf(fmaxf(0.9f * powf(fmaxf(err, 1e-10f), -0.2f), 0.2f), 10.0f);
      if (accept) {
        t = t + dte;
        for (int i = gtid; i < STATE; i += GSZ) {
          yv[i] = y1[i];
          kv[0][i] = kv[6][i];   // FSAL: k1 <- k7
        }
        gbar(sync);
      }
      dtv = dte * factor;
    }
    t = t1;
    // emit: means + eps*stddevs
    for (int i = gtid; i < DATA_DIM; i += GSZ)
      out[(size_t)step * DATA_DIM + i] = yv[i] + eps[(size_t)step * DATA_DIM + i] * yv[DATA_DIM + i];
    // yv only read next interval; prior barriers already ordered it
  }
}

// ---------------- host entry ----------------
extern "C" void kernel_launch(void* const* d_in, const int* in_sizes, int n_in,
                              void* d_out, int out_size, void* d_ws, size_t ws_size,
                              hipStream_t stream) {
  const float* y0    = (const float*)d_in[0];
  const float* W_in  = (const float*)d_in[1];
  const float* b_in  = (const float*)d_in[2];
  const float* W_hid = (const float*)d_in[3];
  const float* b_hid = (const float*)d_in[4];
  const float* W_out = (const float*)d_in[5];
  const float* b_out = (const float*)d_in[6];
  const float* eps   = (const float*)d_in[7];
  (void)in_sizes; (void)n_in; (void)out_size; (void)ws_size;
  char* ws = (char*)d_ws;

  init_sync<<<1, 64, 0, stream>>>((unsigned*)ws);
  pack_kernel<<<512, 256, 0, stream>>>(W_in, HIDDEN, STATE, ws, OFF_WIN);
  for (int l = 0; l < 3; ++l)
    pack_kernel<<<256, 256, 0, stream>>>(W_hid + (size_t)l * HIDDEN * HIDDEN,
                                         HIDDEN, HIDDEN, ws, OFF_WH + (size_t)l * WH_STRIDE);
  pack_kernel<<<512, 256, 0, stream>>>(W_out, STATE, HIDDEN, ws, OFF_WOUT);
  ode_main<<<NWG, TPB, 0, stream>>>(y0, b_in, b_hid, b_out, eps, (float*)d_out, ws);
}
